// PrototypeLayer_60799557042336
// MI455X (gfx1250) — matrix operand order repaired and checked by reading, not compiled, not run on hardware
//
#include <hip/hip_runtime.h>

typedef __attribute__((ext_vector_type(16))) __bf16 v16bf;
typedef __attribute__((ext_vector_type(8)))  __bf16 v8bf;
typedef __attribute__((ext_vector_type(8)))  float  v8f;

#define B_DIM 16384
#define P_DIM 4096
#define D_DIM 256

// ---------------------------------------------------------------------------
// fp32 -> bf16 with round-to-nearest-even (bit trick; avoids relying on
// implicit __bf16 conversion support).
__device__ __forceinline__ __bf16 f32_to_bf16(float f) {
    unsigned u = __builtin_bit_cast(unsigned, f);
    u = (u + 0x7FFFu + ((u >> 16) & 1u)) >> 16;
    unsigned short h = (unsigned short)u;
    return __builtin_bit_cast(__bf16, h);
}

// ---------------------------------------------------------------------------
// Pack: fp32 rows -> bf16 rows + squared L2 norm per row.
// One wave (32 lanes) per row of D=256; 8 waves (256 threads) per block.
__global__ __launch_bounds__(256)
void pack_rows_kernel(const float* __restrict__ in,
                      __bf16* __restrict__ outb,
                      float* __restrict__ sq) {
    const int wave = threadIdx.x >> 5;
    const int lane = threadIdx.x & 31;
    const int row  = blockIdx.x * 8 + wave;

    const float* r = in   + (size_t)row * D_DIM;
    __bf16*      o = outb + (size_t)row * D_DIM;

    float s = 0.0f;
#pragma unroll
    for (int i = 0; i < D_DIM / 32; ++i) {
        const int c = lane + i * 32;          // coalesced 128B per iteration
        float v = r[c];
        s += v * v;
        o[c] = f32_to_bf16(v);
    }
    // wave32 butterfly reduction
#pragma unroll
    for (int off = 16; off > 0; off >>= 1)
        s += __shfl_xor(s, off, 32);
    if (lane == 0) sq[row] = s;
}

// ---------------------------------------------------------------------------
// Distance GEMM: out[m,n] = sqrt(max(xsq[m] + psq[n] - 2 * <x_m, p_n>, 0))
// Workgroup: 256 threads = 8 waves, tile 128(M) x 128(N).
// Wave (wm in 0..1, wn in 0..3): 64(M) x 32(N) = 4 x 2 WMMA 16x16 tiles.
__global__ __launch_bounds__(256)
void dist_wmma_kernel(const __bf16* __restrict__ xb,
                      const __bf16* __restrict__ pb,
                      const float* __restrict__ xsq,
                      const float* __restrict__ psq,
                      float* __restrict__ out) {
    const int lane = threadIdx.x & 31;
    const int wave = threadIdx.x >> 5;
    const int wm   = wave & 1;          // 0..1 : M strip of 64
    const int wn   = wave >> 1;         // 0..3 : N strip of 32

    const int mtile = blockIdx.y * 128 + wm * 64;
    const int ntile = blockIdx.x * 128 + wn * 32;

    const int lmod = lane & 15;
    const int lhi  = lane >> 4;         // 0 or 1 (half-wave)

    v8f acc[4][2] = {};                 // fp32 accumulators, 8 VGPRs each

    // A fragment (16x32 bf16): lane holds row M=lmod,
    //   K = lhi*16 + e, e = 0..15 (contiguous, matching B).
    const __bf16* aBase = xb + (size_t)(mtile + lmod) * D_DIM + lhi * 16;
    // B fragment (32x16 bf16, same K order as A): lane holds col N=lmod,
    //   K = lhi*16 + e, e = 0..15 (contiguous).
    const __bf16* bBase = pb + (size_t)(ntile + lmod) * D_DIM + lhi * 16;

    for (int k0 = 0; k0 < D_DIM; k0 += 32) {
        v16bf a[4];
        v16bf b[2];
#pragma unroll
        for (int msub = 0; msub < 4; ++msub) {
            const __bf16* p = aBase + (size_t)(msub * 16) * D_DIM + k0;
            union { v16bf v; v8bf h[2]; } u;
            u.h[0] = *(const v8bf*)(p);        // K = kb .. kb+7
            u.h[1] = *(const v8bf*)(p + 8);    // K = kb+8 .. kb+15
            a[msub] = u.v;
        }
#pragma unroll
        for (int nsub = 0; nsub < 2; ++nsub) {
            const __bf16* p = bBase + (size_t)(nsub * 16) * D_DIM + k0;
            b[nsub] = *(const v16bf*)(p);      // 32B contiguous
        }
#pragma unroll
        for (int msub = 0; msub < 4; ++msub)
#pragma unroll
            for (int nsub = 0; nsub < 2; ++nsub)
                acc[msub][nsub] = __builtin_amdgcn_wmma_f32_16x16x32_bf16(
                    false, a[msub], false, b[nsub],
                    (short)0, acc[msub][nsub], false, false);
    }

    // Epilogue: C tile layout — VGPR r, lanes 0-15: M=r, lanes 16-31: M=8+r,
    // N = lane%16. Fuse norms + sqrt; stream output with non-temporal stores.
#pragma unroll
    for (int nsub = 0; nsub < 2; ++nsub) {
        const int n  = ntile + nsub * 16 + lmod;
        const float pn = psq[n];
#pragma unroll
        for (int msub = 0; msub < 4; ++msub) {
            const int mb = mtile + msub * 16 + lhi * 8;
#pragma unroll
            for (int r = 0; r < 8; ++r) {
                const int m = mb + r;
                float d2 = xsq[m] + pn - 2.0f * acc[msub][nsub][r];
                float d  = __builtin_sqrtf(fmaxf(d2, 0.0f));
                __builtin_nontemporal_store(d, out + (size_t)m * P_DIM + n);
            }
        }
    }
}

// ---------------------------------------------------------------------------
extern "C" void kernel_launch(void* const* d_in, const int* in_sizes, int n_in,
                              void* d_out, int out_size, void* d_ws, size_t ws_size,
                              hipStream_t stream) {
    (void)in_sizes; (void)n_in; (void)out_size; (void)ws_size;

    const float* x = (const float*)d_in[0];        // [B, D] fp32
    const float* p = (const float*)d_in[1];        // [P, D] fp32
    float*       out = (float*)d_out;              // [B, P] fp32

    // Workspace layout (all offsets 256B-aligned):
    //   xb  : B*D bf16  (8 MiB)
    //   pb  : P*D bf16  (2 MiB)
    //   xsq : B  fp32   (64 KiB)
    //   psq : P  fp32   (16 KiB)
    char* ws = (char*)d_ws;
    __bf16* xb  = (__bf16*)ws;
    __bf16* pb  = (__bf16*)(ws + (size_t)B_DIM * D_DIM * sizeof(__bf16));
    float*  xsq = (float*)(ws + (size_t)B_DIM * D_DIM * sizeof(__bf16)
                              + (size_t)P_DIM * D_DIM * sizeof(__bf16));
    float*  psq = xsq + B_DIM;

    pack_rows_kernel<<<B_DIM / 8, 256, 0, stream>>>(x, xb, xsq);
    pack_rows_kernel<<<P_DIM / 8, 256, 0, stream>>>(p, pb, psq);

    dim3 grid(P_DIM / 128, B_DIM / 128);   // (32, 128)
    dist_wmma_kernel<<<grid, 256, 0, stream>>>(xb, pb, xsq, psq, out);
}
